// ContextTargetPredictor_32255204393746
// MI455X (gfx1250) — compile-verified, matrix operand order checked by reading, not run on hardware
//
#include <hip/hip_runtime.h>

// ---------------------------------------------------------------------------
// GNN: h1 = scatter_add(x@W1) + b1 ; GraphNorm ; tanh ; h2 = scatter_add(.@W2)+b2 ; tanh
// N=50000 (= 3125 * 16), D=256, E=312500.
// GEMMs use split-bf16 (hi/lo) V_WMMA_F32_16X16X32_BF16 (3 WMMAs per k-step)
// for ~fp32 accuracy at bf16 matrix-core rate.
// ---------------------------------------------------------------------------

#define GN_N 50000
#define GN_D 256
#define GN_E 312500
#define NB_STATS 250          // stats partial blocks (200 rows each)

typedef __attribute__((ext_vector_type(16))) __bf16 v16bf;
typedef __attribute__((ext_vector_type(8)))  __bf16 v8bf;
typedef __attribute__((ext_vector_type(8)))  float  v8f;

// ---- bf16 helpers ---------------------------------------------------------

__device__ __forceinline__ float bf16_to_f32(__bf16 h) {
    unsigned u = ((unsigned)__builtin_bit_cast(unsigned short, h)) << 16;
    return __builtin_bit_cast(float, u);
}

__device__ __forceinline__ __bf16 f32_to_bf16_rne(float f) {
    unsigned u = __builtin_bit_cast(unsigned, f);
    unsigned short s = (unsigned short)((u + 0x7fffu + ((u >> 16) & 1u)) >> 16);
    return __builtin_bit_cast(__bf16, s);
}

// hi = truncate-to-bf16(f) (exact subtraction), lo = rne-bf16(f - hi)
__device__ __forceinline__ void split_bf16(float f, __bf16& hi, __bf16& lo) {
    unsigned u  = __builtin_bit_cast(unsigned, f);
    unsigned short ht = (unsigned short)(u >> 16);
    hi = __builtin_bit_cast(__bf16, ht);
    float hf = __builtin_bit_cast(float, ((unsigned)ht) << 16);
    lo = f32_to_bf16_rne(f - hf);
}

// Load one 16-bit WMMA fragment slice for this lane: 8 contiguous bf16 at p
// and 8 contiguous bf16 at p+16 (K layout per CDNA5 ISA 7.12.2).
__device__ __forceinline__ v16bf load_frag(const __bf16* __restrict__ p) {
    v8bf a = *(const v8bf*)(p);
    v8bf b = *(const v8bf*)(p + 16);
    v16bf r;
#pragma unroll
    for (int i = 0; i < 8; ++i) { r[i] = a[i]; r[i + 8] = b[i]; }
    return r;
}

// ---- prep kernels ---------------------------------------------------------

// Split + transpose both weight matrices: Wt[n*256+k] = split(W[k*256+n])
__global__ __launch_bounds__(256)
void prep_weights_kernel(const float* __restrict__ W1, const float* __restrict__ W2,
                         __bf16* __restrict__ W1th, __bf16* __restrict__ W1tl,
                         __bf16* __restrict__ W2th, __bf16* __restrict__ W2tl)
{
    int id = blockIdx.x * 256 + threadIdx.x;       // 0 .. 2*65536-1
    int which = id >> 16;
    int idx   = id & 65535;
    int k = idx >> 8;
    int n = idx & 255;
    const float* W = which ? W2 : W1;
    __bf16* th = which ? W2th : W1th;
    __bf16* tl = which ? W2tl : W1tl;
    __bf16 hi, lo;
    split_bf16(W[idx], hi, lo);
    th[n * 256 + k] = hi;
    tl[n * 256 + k] = lo;
}

__global__ __launch_bounds__(256)
void split_x_kernel(const float* __restrict__ x,
                    __bf16* __restrict__ xh, __bf16* __restrict__ xl)
{
    size_t id = (size_t)blockIdx.x * 256 + threadIdx.x;   // N*D threads
    __bf16 hi, lo;
    split_bf16(x[id], hi, lo);
    xh[id] = hi;
    xl[id] = lo;
}

// ---- split-bf16 WMMA GEMM: C[N,256] = A[N,256] @ W[256,256] ---------------
// A given as hi/lo bf16 row-major; W given pre-transposed (Bt[n][k]) hi/lo.
// 128 threads = 4 waves; block owns one 16-row strip (tileM = blockIdx.x);
// each wave computes 4 adjacent 16x16 output tiles, reusing its A fragments.

__global__ __launch_bounds__(128)
void gemm_bf16x2_kernel(const __bf16* __restrict__ Ah, const __bf16* __restrict__ Al,
                        const __bf16* __restrict__ Bth, const __bf16* __restrict__ Btl,
                        float* __restrict__ C)
{
    const int lane  = threadIdx.x & 31;
    const int wid   = threadIdx.x >> 5;          // 0..3
    const int tileM = blockIdx.x;                // 0..3124
    const int col   = lane & 15;                 // N-col within tile / A row
    const int kb    = (lane >> 4) * 8;           // lane-half K base (0 or 8)

    const __bf16* arow_h = Ah + (size_t)(tileM * 16 + col) * 256;
    const __bf16* arow_l = Al + (size_t)(tileM * 16 + col) * 256;

    v8f acc[4];
#pragma unroll
    for (int t = 0; t < 4; ++t)
#pragma unroll
        for (int j = 0; j < 8; ++j) acc[t][j] = 0.0f;

#pragma unroll 2
    for (int k0 = 0; k0 < 256; k0 += 32) {
        v16bf ah = load_frag(arow_h + k0 + kb);
        v16bf al = load_frag(arow_l + k0 + kb);
#pragma unroll
        for (int t = 0; t < 4; ++t) {
            const int n = wid * 64 + t * 16 + col;
            v16bf bh = load_frag(Bth + (size_t)n * 256 + k0 + kb);
            v16bf bl = load_frag(Btl + (size_t)n * 256 + k0 + kb);
            // A@B ~= Ah@Bh + Ah@Bl + Al@Bh  (fp32 accumulate)
            acc[t] = __builtin_amdgcn_wmma_f32_16x16x32_bf16(
                         false, ah, false, bh, (short)0, acc[t], false, false);
            acc[t] = __builtin_amdgcn_wmma_f32_16x16x32_bf16(
                         false, ah, false, bl, (short)0, acc[t], false, false);
            acc[t] = __builtin_amdgcn_wmma_f32_16x16x32_bf16(
                         false, al, false, bh, (short)0, acc[t], false, false);
        }
    }

    // C/D layout: lane col = lane&15; VGPR j -> row j + 8*(lane>=16)
    const int rowoff = (lane >> 4) * 8;
#pragma unroll
    for (int t = 0; t < 4; ++t) {
        const int n = wid * 64 + t * 16 + col;
#pragma unroll
        for (int j = 0; j < 8; ++j)
            C[(size_t)(tileM * 16 + rowoff + j) * 256 + n] = acc[t][j];
    }
}

// ---- graph ops ------------------------------------------------------------

__global__ __launch_bounds__(256)
void init_bias_kernel(float* __restrict__ out, const float* __restrict__ b)
{
    size_t id = (size_t)blockIdx.x * 256 + threadIdx.x;   // N*D threads
    out[id] = b[id & 255];
}

// 64 threads per edge, float4 per thread; agg target is L2-resident.
__global__ __launch_bounds__(256)
void scatter_add_kernel(const float* __restrict__ h, const int* __restrict__ ei,
                        float* __restrict__ out)
{
    long long id = (long long)blockIdx.x * 256 + threadIdx.x;  // E*64 ids
    if (id >= (long long)GN_E * 64) return;
    int e    = (int)(id >> 6);
    int part = (int)(id & 63);
    int src = ei[e];
    int dst = ei[GN_E + e];
    const float4 v = *(const float4*)(h + (size_t)src * 256 + part * 4);
    float* o = out + (size_t)dst * 256 + part * 4;
    atomicAdd(o + 0, v.x);
    atomicAdd(o + 1, v.y);
    atomicAdd(o + 2, v.z);
    atomicAdd(o + 3, v.w);
}

// Deterministic column stats stage 1: block b sums rows [b*200,(b+1)*200).
__global__ __launch_bounds__(256)
void col_partial_kernel(const float* __restrict__ agg, float* __restrict__ part)
{
    const int d  = threadIdx.x;
    const int r0 = blockIdx.x * 200;
    float s1 = 0.0f, s2 = 0.0f;
    for (int r = r0; r < r0 + 200; ++r) {
        float v = agg[(size_t)r * 256 + d];
        s1 += v;
        s2 += v * v;
    }
    part[blockIdx.x * 256 + d]                 = s1;
    part[NB_STATS * 256 + blockIdx.x * 256 + d] = s2;
}

// Stage 2: fixed-order reduction; fold GraphNorm into per-feature scale/shift.
__global__ __launch_bounds__(256)
void col_final_kernel(const float* __restrict__ part,
                      const float* __restrict__ gw, const float* __restrict__ gb,
                      const float* __restrict__ gms,
                      float* __restrict__ scale, float* __restrict__ shift)
{
    const int d = threadIdx.x;
    float s1 = 0.0f, s2 = 0.0f;
    for (int b = 0; b < NB_STATS; ++b) {
        s1 += part[b * 256 + d];
        s2 += part[NB_STATS * 256 + b * 256 + d];
    }
    const float invN = 1.0f / (float)GN_N;
    float mean = s1 * invN;
    float ms   = mean * gms[d];
    float var  = s2 * invN - 2.0f * ms * mean + ms * ms;  // E[(x-ms)^2]
    float inv  = rsqrtf(var + 1e-5f);
    float sc   = gw[d] * inv;
    scale[d] = sc;
    shift[d] = gb[d] - ms * sc;
}

// out = tanh(agg*scale + shift), split to bf16 hi/lo for the next GEMM.
__global__ __launch_bounds__(256)
void norm_tanh_split_kernel(const float* __restrict__ agg,
                            const float* __restrict__ scale,
                            const float* __restrict__ shift,
                            __bf16* __restrict__ th, __bf16* __restrict__ tl)
{
    size_t id = (size_t)blockIdx.x * 256 + threadIdx.x;   // N*D threads
    int d = (int)(id & 255);
    float t = tanhf(fmaf(agg[id], scale[d], shift[d]));
    __bf16 hi, lo;
    split_bf16(t, hi, lo);
    th[id] = hi;
    tl[id] = lo;
}

__global__ __launch_bounds__(256)
void tanh_inplace_kernel(float* __restrict__ out)
{
    size_t id = (size_t)blockIdx.x * 256 + threadIdx.x;   // N*D threads
    out[id] = tanhf(out[id]);
}

// ---------------------------------------------------------------------------

extern "C" void kernel_launch(void* const* d_in, const int* in_sizes, int n_in,
                              void* d_out, int out_size, void* d_ws, size_t ws_size,
                              hipStream_t stream)
{
    (void)in_sizes; (void)n_in; (void)out_size; (void)ws_size;

    const float* x   = (const float*)d_in[0];
    const int*   ei  = (const int*)  d_in[1];   // [2, E]: row0=src, row1=dst
    const float* W1  = (const float*)d_in[2];
    const float* b1  = (const float*)d_in[3];
    const float* W2  = (const float*)d_in[4];
    const float* b2  = (const float*)d_in[5];
    const float* gw  = (const float*)d_in[6];
    const float* gb  = (const float*)d_in[7];
    const float* gms = (const float*)d_in[8];
    float* out = (float*)d_out;

    // Workspace carve-up (all offsets 256B aligned).
    char* ws = (char*)d_ws;
    const size_t ND = (size_t)GN_N * GN_D;            // 12.8M
    __bf16* xh   = (__bf16*)(ws);                     // 25.6 MB
    __bf16* xl   = (__bf16*)(ws + ND * 2);            // 25.6 MB
    float*  h    = (float*) (ws + ND * 4);            // 51.2 MB
    char*   p    = ws + ND * 4 + ND * 4;
    __bf16* W1th = (__bf16*)(p);            p += 65536 * 2;
    __bf16* W1tl = (__bf16*)(p);            p += 65536 * 2;
    __bf16* W2th = (__bf16*)(p);            p += 65536 * 2;
    __bf16* W2tl = (__bf16*)(p);            p += 65536 * 2;
    float*  part = (float*)(p);             p += (size_t)NB_STATS * 256 * 2 * 4;
    float*  scal = (float*)(p);             p += 256 * 4;
    float*  shif = (float*)(p);

    const int ND_BLOCKS = (int)(ND / 256);            // 50000
    const int SC_BLOCKS = (int)(((long long)GN_E * 64 + 255) / 256); // 78125
    const int GEMM_BLOCKS = GN_N / 16;                // 3125

    // Prep: split weights (transposed) and x into bf16 hi/lo.
    prep_weights_kernel<<<512, 256, 0, stream>>>(W1, W2, W1th, W1tl, W2th, W2tl);
    split_x_kernel<<<ND_BLOCKS, 256, 0, stream>>>(x, xh, xl);

    // Layer 1: h = x @ W1 (WMMA), agg into d_out (init with b1), GraphNorm+tanh.
    gemm_bf16x2_kernel<<<GEMM_BLOCKS, 128, 0, stream>>>(xh, xl, W1th, W1tl, h);
    init_bias_kernel<<<ND_BLOCKS, 256, 0, stream>>>(out, b1);
    scatter_add_kernel<<<SC_BLOCKS, 256, 0, stream>>>(h, ei, out);
    col_partial_kernel<<<NB_STATS, 256, 0, stream>>>(out, part);
    col_final_kernel<<<1, 256, 0, stream>>>(part, gw, gb, gms, scal, shif);
    norm_tanh_split_kernel<<<ND_BLOCKS, 256, 0, stream>>>(out, scal, shif, xh, xl);

    // Layer 2: h2 = t @ W2 (WMMA), agg into d_out (init with b2), tanh.
    gemm_bf16x2_kernel<<<GEMM_BLOCKS, 128, 0, stream>>>(xh, xl, W2th, W2tl, h);
    init_bias_kernel<<<ND_BLOCKS, 256, 0, stream>>>(out, b2);
    scatter_add_kernel<<<SC_BLOCKS, 256, 0, stream>>>(h, ei, out);
    tanh_inplace_kernel<<<ND_BLOCKS, 256, 0, stream>>>(out);
}
